// NemoEncoder_41205916238450
// MI455X (gfx1250) — matrix operand order repaired and checked by reading, not compiled, not run on hardware
//
#include <hip/hip_runtime.h>
#include <math.h>

// ---------------- constants ----------------
#define B_   16
#define FEAT 80
#define TIME 1600
#define D_   512
#define H_   8
#define NL_  16
#define DFF_ 2048
#define CCH_ 256
#define KW_  31
#define DH_  64
#define T1_  800   // after conv1 (stride2)
#define F1_  40
#define T2_  400   // after conv2
#define F2_  20
#define M_   (B_*T2_)     // 6400 rows
#define KSUB (CCH_*F2_)   // 5120
#define P_   (2*T2_-1)    // 799 rel positions

typedef _Float16 h16;
typedef __attribute__((ext_vector_type(8)))  _Float16 v8h;
typedef __attribute__((ext_vector_type(16))) _Float16 v16h;
typedef __attribute__((ext_vector_type(8)))  float    v8f;
typedef __attribute__((ext_vector_type(4)))  float    v4f;
typedef __attribute__((ext_vector_type(4)))  unsigned int u32x4;
typedef __attribute__((ext_vector_type(8)))  int      i32x8;
typedef __attribute__((ext_vector_type(4)))  int      i32x4;

static __device__ inline v8f wmma32(v16h a, v16h b, v8f c) {
  return __builtin_amdgcn_wmma_f32_16x16x32_f16(false, a, false, b, (short)0, c, false, false);
}

// A fragment (16x32 f16, MxK) from LDS. lane m = lane&15; sel = lane>>4.
// halves 0..7  -> K = koff + sel*16 + i ; halves 8..15 -> K = koff + sel*16 + 8 + i
static __device__ inline v16h fragA_lds(const h16* s, int stride, int koff, int lane) {
  int m = lane & 15, sel = lane >> 4;
  const h16* p = s + m * stride + koff;
  v16h r;
#pragma unroll
  for (int i = 0; i < 8; ++i) { r[i] = p[sel * 16 + i]; r[8 + i] = p[sel * 16 + 8 + i]; }
  return r;
}

// B fragment (32x16, KxN) from f16 weight matrix W[N][ld] (row-major, K contiguous).
static __device__ inline v16h fragB_h16(const h16* W, int ld, int nbase, int kb, int lane) {
  const h16* p = W + (size_t)(nbase + (lane & 15)) * ld + kb + (lane >> 4) * 16;
  v8h a = *(const v8h*)p;
  v8h b = *(const v8h*)(p + 8);
  v16h r;
#pragma unroll
  for (int i = 0; i < 8; ++i) { r[i] = a[i]; r[8 + i] = b[i]; }
  return r;
}

static __device__ inline float siluf(float x) { return x / (1.0f + expf(-x)); }

// ---- Tensor Data Mover: async 2-D tile (rows x 32 halfs) global -> LDS ----
// D# group0: count=1, lds_addr, global_addr(57b), type=2. group1: data_size=2B,
// pad_enable, pad_interval=16 DWORDs (row of 32 halfs), pad_amount=4 DWORDs
// (row stride 40 halfs in LDS), tensor dims clamped so OOB rows read zero.
static __device__ inline void tdm_load_tile(const h16* gsrc, unsigned ldsAddr,
                                            int kRemain, int rows, int ldK) {
  unsigned long long ga = (unsigned long long)(uintptr_t)gsrc;
  u32x4 g0;
  g0[0] = 1u;                                   // count=1, user mode
  g0[1] = ldsAddr;                              // lds_addr
  g0[2] = (unsigned)ga;                         // global_addr[31:0]
  g0[3] = (unsigned)((ga >> 32) & 0x01FFFFFFu) | (2u << 30);  // addr[56:32] | type=2
  unsigned td0 = (unsigned)kRemain;             // tensor_dim0 (elements)
  unsigned td1 = (unsigned)rows;                // tensor_dim1
  unsigned long long st0 = (unsigned long long)ldK;
  i32x8 g1;
  g1[0] = (int)((1u << 16) | (1u << 20) | (3u << 22) | (3u << 25)); // dsize=2B,pad_en,int=16dw,amt=4dw
  g1[1] = (int)((td0 & 0xFFFFu) << 16);                       // tensor_dim0 lo16 @ bits 63:48
  g1[2] = (int)(((td0 >> 16) & 0xFFFFu) | ((td1 & 0xFFFFu) << 16));
  g1[3] = (int)(((td1 >> 16) & 0xFFFFu) | (32u << 16));       // tile_dim0 = 32
  g1[4] = (int)64;                                            // tile_dim1 = 64 (tile_dim2=0)
  g1[5] = (int)(unsigned)(st0 & 0xFFFFFFFFu);                 // dim0 stride lo32
  g1[6] = (int)(unsigned)((st0 >> 32) & 0xFFFFu);             // dim0 stride hi16
  g1[7] = 0;
  i32x4 z4 = (i32x4)0;
#if __clang_major__ >= 23
  i32x8 z8 = (i32x8)0;
  __builtin_amdgcn_tensor_load_to_lds(g0, g1, z4, z4, z8, 0);
#else
  __builtin_amdgcn_tensor_load_to_lds(g0, g1, z4, z4, 0);
#endif
}

// ---------------- generic GEMM: out = act(A @ W^T + bias) * alpha + resid ----------------
// A: f16 [M,K]. W: f16 [N,K]. Block tile 64x128, 8 waves of 16x64 (4 accums).
// A tile staged into LDS by the Tensor Data Mover, double buffered.
__global__ __launch_bounds__(256) void gemm_kernel(
    const h16* __restrict__ A, const h16* __restrict__ W,
    const float* __restrict__ bias, const float* __restrict__ resid,
    float* __restrict__ outF, h16* __restrict__ outH,
    int M, int N, int K, int act, float alpha)
{
  __shared__ h16 sA[2][64 * 40];
  int tid = threadIdx.x, lane = tid & 31, wid = tid >> 5;
  int bm = blockIdx.y, bn = blockIdx.x;
  int wm = wid >> 1, wn = wid & 1;              // 4 (M) x 2 (N) waves
  v8f acc0 = {}, acc1 = {}, acc2 = {}, acc3 = {};

  int rows = M - bm * 64; if (rows > 64) rows = 64;
  const h16* Abase = A + (size_t)bm * 64 * K;
  unsigned lds0 = (unsigned)(uintptr_t)(void*)&sA[0][0];
  unsigned lds1 = (unsigned)(uintptr_t)(void*)&sA[1][0];

  if (wid == 0) tdm_load_tile(Abase, lds0, K, rows, K);

  int curb = 0;
  for (int kb = 0; kb < K; kb += 32) {
    if (wid == 0) __builtin_amdgcn_s_wait_tensorcnt(0);
    __syncthreads();
    if (wid == 0 && kb + 32 < K)
      tdm_load_tile(Abase + kb + 32, curb ? lds0 : lds1, K - kb - 32, rows, K);

    v16h af = fragA_lds(&sA[curb][wm * 16 * 40], 40, 0, lane);
    int nb = bn * 128 + wn * 64;
    v16h b0 = fragB_h16(W, K, nb,      kb, lane);
    v16h b1 = fragB_h16(W, K, nb + 16, kb, lane);
    v16h b2 = fragB_h16(W, K, nb + 32, kb, lane);
    v16h b3 = fragB_h16(W, K, nb + 48, kb, lane);
    acc0 = wmma32(af, b0, acc0);
    acc1 = wmma32(af, b1, acc1);
    acc2 = wmma32(af, b2, acc2);
    acc3 = wmma32(af, b3, acc3);
    curb ^= 1;
  }

  int sel = lane >> 4;
  int cb = bn * 128 + wn * 64 + (lane & 15);
#pragma unroll
  for (int e = 0; e < 8; ++e) {
    int r = bm * 64 + wm * 16 + sel * 8 + e;
    if (r < M) {
      float v[4] = { acc0[e], acc1[e], acc2[e], acc3[e] };
#pragma unroll
      for (int s = 0; s < 4; ++s) {
        int c = cb + s * 16;
        float x = v[s];
        if (bias) x += bias[c];
        if (act == 1) x = siluf(x);
        x *= alpha;
        size_t o = (size_t)r * N + c;
        if (resid) x += resid[o];
        if (outF) outF[o] = x;
        if (outH) outH[o] = (h16)x;
      }
    }
  }
}

// ---------------- f32 -> f16 weight cast ----------------
__global__ __launch_bounds__(256) void castw_kernel(const float* __restrict__ w,
                                                    h16* __restrict__ o, int n)
{
  int idx = (blockIdx.x * 256 + threadIdx.x) * 4;
  if (idx + 3 < n) {
    v4f x = *(const v4f*)(w + idx);
#pragma unroll
    for (int i = 0; i < 4; ++i) o[idx + i] = (h16)x[i];
  } else {
    for (int i = idx; i < n; ++i) o[i] = (h16)w[i];
  }
}

// ---------------- layernorm over D=512 (wave per row) ----------------
__global__ __launch_bounds__(256) void ln_kernel(
    const float* __restrict__ x, const float* __restrict__ g, const float* __restrict__ b,
    float* __restrict__ outF, h16* __restrict__ outH, int M)
{
  int lane = threadIdx.x & 31, wid = threadIdx.x >> 5;
  int row = blockIdx.x * 8 + wid;
  if (row >= M) return;
  const float* xr = x + (size_t)row * D_;
  float v[16]; float s = 0.f;
#pragma unroll
  for (int i = 0; i < 16; ++i) { v[i] = xr[lane + i * 32]; s += v[i]; }
  for (int o = 16; o > 0; o >>= 1) s += __shfl_xor(s, o, 32);
  float mean = s * (1.0f / D_);
  float sq = 0.f;
#pragma unroll
  for (int i = 0; i < 16; ++i) { float d = v[i] - mean; sq += d * d; }
  for (int o = 16; o > 0; o >>= 1) sq += __shfl_xor(sq, o, 32);
  float inv = rsqrtf(sq * (1.0f / D_) + 1e-5f);
#pragma unroll
  for (int i = 0; i < 16; ++i) {
    int d = lane + i * 32;
    float o = (v[i] - mean) * inv * g[d] + b[d];
    if (outF) outF[(size_t)row * D_ + d] = o;
    if (outH) outH[(size_t)row * D_ + d] = (h16)o;
  }
}

// ---------------- subsampling convs ----------------
__global__ __launch_bounds__(256) void conv1_kernel(
    const float* __restrict__ audio, const float* __restrict__ w,
    const float* __restrict__ bias, h16* __restrict__ out)
{
  int idx = blockIdx.x * 256 + threadIdx.x;
  const int total = B_ * CCH_ * T1_ * F1_;
  if (idx >= total) return;
  int fo = idx % F1_, to = (idx / F1_) % T1_, co = (idx / (F1_ * T1_)) % CCH_, b = idx / (F1_ * T1_ * CCH_);
  float acc = bias[co];
#pragma unroll
  for (int kr = 0; kr < 3; ++kr) {
    int ti = 2 * to + kr - 1;
    if (ti < 0 || ti >= TIME) continue;
#pragma unroll
    for (int kc = 0; kc < 3; ++kc) {
      int fi = 2 * fo + kc - 1;
      if (fi < 0 || fi >= FEAT) continue;
      acc += audio[((size_t)b * FEAT + fi) * TIME + ti] * w[co * 9 + kr * 3 + kc];
    }
  }
  out[idx] = (h16)fmaxf(acc, 0.f);
}

__global__ __launch_bounds__(256) void conv2_kernel(
    const h16* __restrict__ c1, const float* __restrict__ w,
    const float* __restrict__ bias, h16* __restrict__ asub)
{
  int idx = blockIdx.x * 256 + threadIdx.x;
  const int total = B_ * CCH_ * T2_ * F2_;
  if (idx >= total) return;
  int fo = idx % F2_, to = (idx / F2_) % T2_, co = (idx / (F2_ * T2_)) % CCH_, b = idx / (F2_ * T2_ * CCH_);
  float acc = bias[co];
  for (int ci = 0; ci < CCH_; ++ci) {
    const h16* cp = c1 + ((size_t)(b * CCH_ + ci) * T1_) * F1_;
    const float* wp = w + ((size_t)co * CCH_ + ci) * 9;
#pragma unroll
    for (int kr = 0; kr < 3; ++kr) {
      int ti = 2 * to + kr - 1;
      if (ti < 0 || ti >= T1_) continue;
#pragma unroll
      for (int kc = 0; kc < 3; ++kc) {
        int fi = 2 * fo + kc - 1;
        if (fi < 0 || fi >= F1_) continue;
        acc += (float)cp[ti * F1_ + fi] * wp[kr * 3 + kc];
      }
    }
  }
  asub[((size_t)(b * T2_ + to)) * KSUB + co * F2_ + fo] = (h16)fmaxf(acc, 0.f);
}

// ---------------- positional encoding ----------------
__global__ __launch_bounds__(256) void pe_kernel(h16* __restrict__ pe)
{
  int idx = blockIdx.x * 256 + threadIdx.x;
  if (idx >= P_ * (D_ / 2)) return;
  int p = idx / (D_ / 2), i = idx % (D_ / 2);
  float pos = (float)(T2_ - 1 - p);
  float div = expf((float)(2 * i) * (-logf(10000.0f) / (float)D_));
  float a = pos * div;
  pe[(size_t)p * D_ + 2 * i]     = (h16)sinf(a);
  pe[(size_t)p * D_ + 2 * i + 1] = (h16)cosf(a);
}

// ---------------- subsampled lengths ----------------
__global__ void lsub_kernel(const int* __restrict__ len, int* __restrict__ lsub)
{
  int b = threadIdx.x;
  if (b < B_) {
    int l = len[b];
    l = (l - 1) / 2 + 1;
    l = (l - 1) / 2 + 1;
    lsub[b] = l;
  }
}

// ---------------- GLU + pad mask ----------------
__global__ __launch_bounds__(256) void glu_kernel(
    const float* __restrict__ pw, const int* __restrict__ lsub, float* __restrict__ out)
{
  int idx = blockIdx.x * 256 + threadIdx.x;
  if (idx >= M_ * D_) return;
  int d = idx % D_;
  int row = idx / D_;
  int t = row % T2_, b = row / T2_;
  float a = pw[(size_t)row * (2 * D_) + d];
  float g = pw[(size_t)row * (2 * D_) + D_ + d];
  float v = a * (1.0f / (1.0f + expf(-g)));
  if (t >= lsub[b]) v = 0.f;
  out[idx] = v;
}

// ---------------- depthwise conv (K=31) + BN(eval) + SiLU ----------------
__global__ __launch_bounds__(256) void dwconv_kernel(
    const float* __restrict__ x, const float* __restrict__ w, const float* __restrict__ bias,
    const float* __restrict__ bng, const float* __restrict__ bnb, h16* __restrict__ out)
{
  int idx = blockIdx.x * 256 + threadIdx.x;
  if (idx >= M_ * D_) return;
  int d = idx % D_;
  int row = idx / D_;
  int t = row % T2_, b = row / T2_;
  float acc = bias[d];
  const float* wd = w + d * KW_;
#pragma unroll 4
  for (int k = 0; k < KW_; ++k) {
    int tt = t + k - KW_ / 2;
    if (tt >= 0 && tt < T2_)
      acc += x[((size_t)(b * T2_ + tt)) * D_ + d] * wd[k];
  }
  acc = acc * (bng[d] * rsqrtf(1.0f + 1e-5f)) + bnb[d];
  out[idx] = (h16)siluf(acc);
}

// ---------------- rel-pos attention (one block per (b,h,16 q rows)) ----------------
#define SSTR 416
__global__ __launch_bounds__(256) void attn_kernel(
    const h16* __restrict__ qh, const h16* __restrict__ kh, const h16* __restrict__ vh,
    const h16* __restrict__ pmh, const float* __restrict__ ub, const float* __restrict__ vb,
    h16* __restrict__ ah)
{
  __shared__ float sS[16 * SSTR];
  __shared__ h16  sP[16 * SSTR];
  __shared__ h16  sQu[16 * 72];
  __shared__ h16  sQv[16 * 72];
  int tid = threadIdx.x, lane = tid & 31, wid = tid >> 5;
  int blk = blockIdx.x;
  int qt = blk % (T2_ / 16);
  int h  = (blk / (T2_ / 16)) % H_;
  int b  = blk / ((T2_ / 16) * H_);
  int q0 = qt * 16;

  for (int e = tid; e < 16 * DH_; e += 256) {
    int r = e >> 6, d = e & 63;
    float qv = (float)qh[((size_t)(b * T2_ + q0 + r)) * D_ + h * DH_ + d];
    sQu[r * 72 + d] = (h16)(qv + ub[h * DH_ + d]);
    sQv[r * 72 + d] = (h16)(qv + vb[h * DH_ + d]);
  }
  __syncthreads();

  // ac = (q+u) @ K^T
  for (int kt = wid; kt < T2_ / 16; kt += 8) {
    v8f acc = {};
#pragma unroll
    for (int c = 0; c < 2; ++c) {
      v16h af = fragA_lds(sQu, 72, c * 32, lane);
      int kidx = kt * 16 + (lane & 15);
      const h16* p = kh + ((size_t)(b * T2_ + kidx)) * D_ + h * DH_ + c * 32 + (lane >> 4) * 16;
      v8h lo = *(const v8h*)p;
      v8h hi = *(const v8h*)(p + 8);
      v16h bf;
#pragma unroll
      for (int i = 0; i < 8; ++i) { bf[i] = lo[i]; bf[8 + i] = hi[i]; }
      acc = wmma32(af, bf, acc);
    }
    int sel = lane >> 4, col = kt * 16 + (lane & 15);
#pragma unroll
    for (int e = 0; e < 8; ++e) sS[(sel * 8 + e) * SSTR + col] = acc[e];
  }
  __syncthreads();

  // bd[q,k] = (q+v) . pm[T-1-q+k]   (rel-shift folded into the index)
  for (int e = tid; e < 16 * T2_; e += 256) {
    int r = e / T2_, k = e % T2_;
    int p = (T2_ - 1) - (q0 + r) + k;
    const h16* pm = pmh + (size_t)p * D_ + h * DH_;
    const h16* qv = sQv + r * 72;
    float s = 0.f;
#pragma unroll 8
    for (int d = 0; d < DH_; ++d) s += (float)qv[d] * (float)pm[d];
    sS[r * SSTR + k] += s;
  }
  __syncthreads();

  // softmax (scale 1/sqrt(DH)), 2 rows per wave
  const float scale = 0.125f;
#pragma unroll
  for (int rr = 0; rr < 2; ++rr) {
    int r = wid * 2 + rr;
    float mx = -1e30f;
    for (int k = lane; k < T2_; k += 32) mx = fmaxf(mx, sS[r * SSTR + k] * scale);
    for (int o = 16; o > 0; o >>= 1) mx = fmaxf(mx, __shfl_xor(mx, o, 32));
    float sum = 0.f;
    for (int k = lane; k < T2_; k += 32) {
      float pz = expf(sS[r * SSTR + k] * scale - mx);
      sS[r * SSTR + k] = pz; sum += pz;
    }
    for (int o = 16; o > 0; o >>= 1) sum += __shfl_xor(sum, o, 32);
    float inv = 1.0f / sum;
    for (int k = lane; k < T2_; k += 32) sP[r * SSTR + k] = (h16)(sS[r * SSTR + k] * inv);
    for (int k = T2_ + lane; k < SSTR; k += 32) sP[r * SSTR + k] = (h16)0.f;
  }
  __syncthreads();

  // out = P @ V   (4 waves x 16 d-cols, K padded to 416 with zero probs)
  if (wid < 4) {
    v8f acc = {};
    for (int kc = 0; kc < SSTR / 32; ++kc) {
      v16h af = fragA_lds(sP, SSTR, kc * 32, lane);
      v16h bf;
      int n = lane & 15, sel = lane >> 4;
#pragma unroll
      for (int i = 0; i < 16; ++i) {
        int kr = kc * 32 + sel * 16 + i;
        if (kr > T2_ - 1) kr = T2_ - 1;
        bf[i] = vh[((size_t)(b * T2_ + kr)) * D_ + h * DH_ + wid * 16 + n];
      }
      acc = wmma32(af, bf, acc);
    }
    int sel = lane >> 4, n = lane & 15;
#pragma unroll
    for (int e = 0; e < 8; ++e) {
      int r = sel * 8 + e;
      ah[((size_t)(b * T2_ + q0 + r)) * D_ + h * DH_ + wid * 16 + n] = (h16)acc[e];
    }
  }
}

// ---------------- final transpose [B,T,D] -> [B,D,T] ----------------
__global__ __launch_bounds__(256) void transpose_kernel(const float* __restrict__ x, float* __restrict__ out)
{
  int idx = blockIdx.x * 256 + threadIdx.x;
  if (idx >= B_ * D_ * T2_) return;
  int t = idx % T2_, d = (idx / T2_) % D_, b = idx / (T2_ * D_);
  out[idx] = x[((size_t)(b * T2_ + t)) * D_ + d];
}

__global__ void lens_out_kernel(const int* __restrict__ lsub, float* __restrict__ out)
{
  int b = threadIdx.x;
  if (b < B_) out[(size_t)B_ * D_ * T2_ + b] = (float)lsub[b];
}

// =================================================================
extern "C" void kernel_launch(void* const* d_in, const int* in_sizes, int n_in,
                              void* d_out, int out_size, void* d_ws, size_t ws_size,
                              hipStream_t stream)
{
  (void)in_sizes; (void)n_in; (void)out_size; (void)ws_size;
  const float* audio     = (const float*)d_in[0];
  const int*   length    = (const int*)d_in[1];
  const float* sub_c1_w  = (const float*)d_in[2];
  const float* sub_c1_b  = (const float*)d_in[3];
  const float* sub_c2_w  = (const float*)d_in[4];
  const float* sub_c2_b  = (const float*)d_in[5];
  const float* sub_lin_w = (const float*)d_in[6];
  const float* sub_lin_b = (const float*)d_in[7];
  const float* pos_u     = (const float*)d_in[8];
  const float* pos_v     = (const float*)d_in[9];
  const float* ln_ff1_g = (const float*)d_in[10];
  const float* ln_ff1_b = (const float*)d_in[11];
  const float* ff1_w1   = (const float*)d_in[12];
  const float* ff1_b1   = (const float*)d_in[13];
  const float* ff1_w2   = (const float*)d_in[14];
  const float* ff1_b2   = (const float*)d_in[15];
  const float* ln_att_g = (const float*)d_in[16];
  const float* ln_att_b = (const float*)d_in[17];
  const float* wq = (const float*)d_in[18];
  const float* bq = (const float*)d_in[19];
  const float* wk = (const float*)d_in[20];
  const float* bk = (const float*)d_in[21];
  const float* wv = (const float*)d_in[22];
  const float* bv = (const float*)d_in[23];
  const float* wo = (const float*)d_in[24];
  const float* bo = (const float*)d_in[25];
  const float* wpos = (const float*)d_in[26];
  const float* ln_conv_g = (const float*)d_in[27];
  const float* ln_conv_b = (const float*)d_in[28];
  const float* pw1_w = (const float*)d_in[29];
  const float* pw1_b = (const float*)d_in[30];
  const float* dw_w  = (const float*)d_in[31];
  const float* dw_b  = (const float*)d_in[32];
  const float* bn_g  = (const float*)d_in[33];
  const float* bn_b  = (const float*)d_in[34];
  const float* pw2_w = (const float*)d_in[35];
  const float* pw2_b = (const float*)d_in[36];
  const float* ln_ff2_g = (const float*)d_in[37];
  const float* ln_ff2_b = (const float*)d_in[38];
  const float* ff2_w1 = (const float*)d_in[39];
  const float* ff2_b1 = (const float*)d_in[40];
  const float* ff2_w2 = (const float*)d_in[41];
  const float* ff2_b2 = (const float*)d_in[42];
  const float* ln_out_g = (const float*)d_in[43];
  const float* ln_out_b = (const float*)d_in[44];

  // ---- workspace carving ----
  char* base = (char*)d_ws;
  size_t off = 0;
  auto alloc = [&](size_t bytes) -> void* {
    off = (off + 255) & ~(size_t)255;
    void* p = base + off;
    off += bytes;
    return p;
  };
  h16*   conv1h = (h16*)  alloc((size_t)B_ * CCH_ * T1_ * F1_ * sizeof(h16));
  h16*   Asub   = (h16*)  alloc((size_t)M_ * KSUB * sizeof(h16));
  float* X      = (float*)alloc((size_t)M_ * D_ * sizeof(float));
  float* X2     = (float*)alloc((size_t)M_ * D_ * sizeof(float));
  h16*   Yh     = (h16*)  alloc((size_t)M_ * D_ * sizeof(h16));
  h16*   Hh     = (h16*)  alloc((size_t)M_ * DFF_ * sizeof(h16));
  h16*   Qh     = (h16*)  alloc((size_t)M_ * D_ * sizeof(h16));
  h16*   Kh     = (h16*)  alloc((size_t)M_ * D_ * sizeof(h16));
  h16*   Vh     = (h16*)  alloc((size_t)M_ * D_ * sizeof(h16));
  h16*   Ah     = (h16*)  alloc((size_t)M_ * D_ * sizeof(h16));
  float* PW     = (float*)alloc((size_t)M_ * 2 * D_ * sizeof(float));
  float* Gb     = (float*)alloc((size_t)M_ * D_ * sizeof(float));
  h16*   Dh     = (h16*)  alloc((size_t)M_ * D_ * sizeof(h16));
  h16*   Peh    = (h16*)  alloc((size_t)(P_ + 1) * D_ * sizeof(h16));
  h16*   Pmh    = (h16*)  alloc((size_t)(P_ + 33) * D_ * sizeof(h16));
  h16*   Wh     = (h16*)  alloc((size_t)KSUB * D_ * sizeof(h16));   // f16 weight scratch (max N*K)
  int*   Lsub   = (int*)  alloc(64);

  auto gemm = [&](const h16* A, const float* Wf, const float* bias, const float* resid,
                  float* oF, h16* oH, int M, int N, int K, int act, float alpha) {
    int n = N * K;
    castw_kernel<<<(n / 4 + 255) / 256, 256, 0, stream>>>(Wf, Wh, n);
    dim3 g(N / 128, (M + 63) / 64);
    gemm_kernel<<<g, 256, 0, stream>>>(A, Wh, bias, resid, oF, oH, M, N, K, act, alpha);
  };
  auto ln = [&](const float* x, const float* g, const float* b, float* oF, h16* oH) {
    ln_kernel<<<(M_ + 7) / 8, 256, 0, stream>>>(x, g, b, oF, oH, M_);
  };

  // ---- subsampling ----
  {
    int t1 = B_ * CCH_ * T1_ * F1_;
    conv1_kernel<<<(t1 + 255) / 256, 256, 0, stream>>>(audio, sub_c1_w, sub_c1_b, conv1h);
    int t2 = B_ * CCH_ * T2_ * F2_;
    conv2_kernel<<<(t2 + 255) / 256, 256, 0, stream>>>(conv1h, sub_c2_w, sub_c2_b, Asub);
    const float XSCALE = sqrtf((float)D_);
    gemm(Asub, sub_lin_w, sub_lin_b, nullptr, X, nullptr, M_, D_, KSUB, 0, XSCALE);
    pe_kernel<<<(P_ * (D_ / 2) + 255) / 256, 256, 0, stream>>>(Peh);
    lsub_kernel<<<1, 64, 0, stream>>>(length, Lsub);
  }

  const int EWB = (M_ * D_ + 255) / 256;

  float* cur = X;
  float* alt = X2;
  for (int L = 0; L < NL_; ++L) {
    size_t oD   = (size_t)L * D_;
    size_t oDD  = (size_t)L * D_ * D_;
    size_t oDF  = (size_t)L * DFF_ * D_;
    size_t oF   = (size_t)L * DFF_;
    size_t oPW  = (size_t)L * 2 * D_ * D_;
    size_t oPWb = (size_t)L * 2 * D_;
    size_t oDW  = (size_t)L * D_ * KW_;

    // FF1 (half residual)
    ln(cur, ln_ff1_g + oD, ln_ff1_b + oD, nullptr, Yh);
    gemm(Yh, ff1_w1 + oDF, ff1_b1 + oF, nullptr, nullptr, Hh, M_, DFF_, D_, 1, 1.0f);
    gemm(Hh, ff1_w2 + oDF, ff1_b2 + oD, cur, alt, nullptr, M_, D_, DFF_, 0, 0.5f);
    { float* t = cur; cur = alt; alt = t; }

    // attention
    ln(cur, ln_att_g + oD, ln_att_b + oD, nullptr, Yh);
    gemm(Yh, wq + oDD, bq + oD, nullptr, nullptr, Qh, M_, D_, D_, 0, 1.0f);
    gemm(Yh, wk + oDD, bk + oD, nullptr, nullptr, Kh, M_, D_, D_, 0, 1.0f);
    gemm(Yh, wv + oDD, bv + oD, nullptr, nullptr, Vh, M_, D_, D_, 0, 1.0f);
    gemm(Peh, wpos + oDD, nullptr, nullptr, nullptr, Pmh, P_, D_, D_, 0, 1.0f);
    attn_kernel<<<B_ * H_ * (T2_ / 16), 256, 0, stream>>>(Qh, Kh, Vh, Pmh, pos_u, pos_v, Ah);
    gemm(Ah, wo + oDD, bo + oD, cur, alt, nullptr, M_, D_, D_, 0, 1.0f);
    { float* t = cur; cur = alt; alt = t; }

    // conv module
    ln(cur, ln_conv_g + oD, ln_conv_b + oD, nullptr, Yh);
    gemm(Yh, pw1_w + oPW, pw1_b + oPWb, nullptr, PW, nullptr, M_, 2 * D_, D_, 0, 1.0f);
    glu_kernel<<<EWB, 256, 0, stream>>>(PW, Lsub, Gb);
    dwconv_kernel<<<EWB, 256, 0, stream>>>(Gb, dw_w + oDW, dw_b + oD, bn_g + oD, bn_b + oD, Dh);
    gemm(Dh, pw2_w + oDD, pw2_b + oD, cur, alt, nullptr, M_, D_, D_, 0, 1.0f);
    { float* t = cur; cur = alt; alt = t; }

    // FF2 (half residual) + final LN (in place)
    ln(cur, ln_ff2_g + oD, ln_ff2_b + oD, nullptr, Yh);
    gemm(Yh, ff2_w1 + oDF, ff2_b1 + oF, nullptr, nullptr, Hh, M_, DFF_, D_, 1, 1.0f);
    gemm(Hh, ff2_w2 + oDF, ff2_b2 + oD, cur, alt, nullptr, M_, D_, DFF_, 0, 0.5f);
    { float* t = cur; cur = alt; alt = t; }
    ln(cur, ln_out_g + oD, ln_out_b + oD, cur, nullptr);
  }

  transpose_kernel<<<(B_ * D_ * T2_ + 255) / 256, 256, 0, stream>>>(cur, (float*)d_out);
  lens_out_kernel<<<1, 64, 0, stream>>>(Lsub, (float*)d_out);
}
